// CLGD_86809878986948
// MI455X (gfx1250) — matrix operand order repaired and checked by reading, not compiled, not run on hardware
//
#include <hip/hip_runtime.h>
#include <hip/hip_bf16.h>

// ---------------------------------------------------------------------------
// Types / helpers
// ---------------------------------------------------------------------------
typedef unsigned short u16;
typedef __attribute__((ext_vector_type(16))) __bf16 v16bf;
typedef __attribute__((ext_vector_type(8)))  float  v8f;

union Frag { uint4 q[2]; v16bf v; };

#define LEAKS 0.2f

__device__ __forceinline__ u16 f2bf(float f) {           // RNE f32->bf16
  unsigned u = __float_as_uint(f);
  unsigned r = (u + 0x7fffu + ((u >> 16) & 1u)) >> 16;
  return (u16)r;
}
__device__ __forceinline__ float bf2f(u16 h) {
  return __uint_as_float(((unsigned)h) << 16);
}

__device__ __forceinline__ void prefetch_bytes(const void* p, int bytes,
                                               int tid, int nthreads) {
  for (int off = tid * 64; off < bytes; off += nthreads * 64)
    __builtin_prefetch((const char*)p + off, 0, 0);     // global_prefetch_b8
}

// Multi-tile WMMA K-loop: one A fragment (LDS) shared across NT B tiles.
// A frag (CDNA5 16-bit A 16x32): lane<16 -> M=lane, elems0-7 = K k0..k0+7,
//   elems8-15 = K k0+16..k0+23; lanes>=16 shift K base by 8.
// B frag: N=lane%16, K in same order as A (elems0-7 K k0+8h.., elems8-15 K k0+16+8h..).
template <int KP, int NT>
__device__ __forceinline__ void wmma_multi(const u16* As, int lda,
                                           const u16* Bg, int ldb,
                                           int n0, int nstride, int lane,
                                           v8f* acc) {
  const int r = lane & 15, hi = lane >> 4;
  const u16* ap   = As + r * lda + hi * 8;
  const u16* brow = Bg + (size_t)(hi * 8) * ldb + n0 + r;
#pragma unroll 2
  for (int k0 = 0; k0 < KP; k0 += 32) {
    Frag a;
    a.q[0] = *(const uint4*)(ap + k0);
    a.q[1] = *(const uint4*)(ap + k0 + 16);
#pragma unroll
    for (int t = 0; t < NT; ++t) {
      Frag b;
      const u16* bp = brow + (size_t)k0 * ldb + t * nstride;
#pragma unroll
      for (int s = 0; s < 16; ++s)
        ((u16*)&b)[s] = bp[(size_t)((s < 8) ? s : s + 8) * ldb];
      acc[t] = __builtin_amdgcn_wmma_f32_16x16x32_bf16(false, a.v, false, b.v,
                                                       (short)0, acc[t],
                                                       false, false);
    }
  }
}

// ---------------------------------------------------------------------------
// Workspace layout (bf16 elements, then f32 newG)
// ---------------------------------------------------------------------------
__device__ __host__ constexpr int OFFW[8]  = {0, 147456, 409600, 671744,
                                              802816, 1064960, 1327104, 1589248};
__device__ __host__ constexpr int OFFWT[8] = {1851392, 1998848, 2260992, 2523136,
                                              2654208, 2916352, 3178496, 3440640};
constexpr int OFF_WR = 3702784;   // 544x256
constexpr int OFF_WZ = 3842048;
constexpr int OFF_WG = 3981312;
constexpr size_t WS_BF16_END_BYTES = 4120576ull * 2ull;

// d_out float offsets: s, dsx, dsh, Gnew, dx, lp
constexpr size_t S_OFF = 0, DSX_OFF = 4096, DSH_OFF = 16384, GNEW_OFF = 16896,
                 DX_OFF = 16794112, LP_OFF = 16794114;

// ---------------------------------------------------------------------------
// Weight conversion kernels (f32 -> padded bf16 [+ transpose])
// ---------------------------------------------------------------------------
__global__ void k_conv_sdf(const float* __restrict__ W, u16* Wp, u16* WTp,
                           int IN, int OUT, int INp, int OUTp) {
  int total = INp * OUTp;
  for (int i = blockIdx.x * 256 + threadIdx.x; i < total; i += gridDim.x * 256) {
    int r = i / OUTp, c = i % OUTp;
    float v = (r < IN && c < OUT) ? W[r * OUT + c] : 0.f;
    u16 h = f2bf(v);
    Wp[i] = h;
    WTp[c * INp + r] = h;
  }
}
__global__ void k_conv_gate(const float* __restrict__ W, u16* Wp,
                            int IN, int OUT, int INp) {
  int total = INp * OUT;
  for (int i = blockIdx.x * 256 + threadIdx.x; i < total; i += gridDim.x * 256) {
    int r = i / OUT, c = i % OUT;
    float v = (r < IN) ? W[r * OUT + c] : 0.f;
    Wp[i] = f2bf(v);
  }
}

// ---------------------------------------------------------------------------
// Kernel A: DeepSDF forward + VJP.  1 WG = 16 points, 8 waves.
// ---------------------------------------------------------------------------
struct SdfArgs {
  const float* X; const float* H;
  const float* W8; const float* b8;
  const float* bias[8];
  const u16* wbf;
  float* out;
};

__device__ __forceinline__ void store_leaky(v8f acc, int col, int hi,
                                            u16 (*dst)[512],
                                            unsigned char (*msk)[512],
                                            const float* bias, int nv) {
  float bv = (col < nv) ? bias[col] : 0.f;
#pragma unroll
  for (int j = 0; j < 8; ++j) {
    int m = hi * 8 + j;
    float v = acc[j] + bv;
    bool pos = v > 0.f;
    float o = pos ? v : LEAKS * v;
    if (col >= nv) { o = 0.f; pos = false; }
    dst[m][col] = f2bf(o);
    msk[m][col] = pos ? 1 : 0;
  }
}

__global__ __launch_bounds__(256) void k_sdf(SdfArgs a) {
  __shared__ u16  sInp[16][288];              //  9 216 B (H|X, zero-padded)
  __shared__ u16  sAct[2][16][512];           // 32 768 B (ping-pong acts/grads)
  __shared__ unsigned char sMask[8][16][512]; // 65 536 B (leaky masks)
  __shared__ float sGin[16][288];             // 18 432 B (grad wrt raw input)

  const int tid = threadIdx.x, wid = tid >> 5, lane = tid & 31;
  const int r = lane & 15, hi = lane >> 4;
  const int wg = blockIdx.x, b = wg >> 7, nbase = (wg & 127) * 16;
  const int p16 = tid >> 4, c16 = tid & 15;   // division-free LDS fills

  // input: cols 0..255 = H, 256..258 = X, 259..287 = 0
  for (int c = c16; c < 288; c += 16) {
    float v = 0.f;
    if (c < 256)      v = a.H[b * 256 + c];
    else if (c < 259) v = a.X[(size_t)(b * 2048 + nbase + p16) * 3 + (c - 256)];
    sInp[p16][c] = f2bf(v);
  }
  __syncthreads();

  const int Nv[8] = {512,512,512,253,512,512,512,512};

  // ---------------- forward ----------------
  for (int l = 0; l < 8; ++l) {
    const u16* As  = (l == 0) ? &sInp[0][0] : &sAct[(l + 1) & 1][0][0];
    const int  lda = (l == 0) ? 288 : 512;
    u16 (*dst)[512] = sAct[l & 1];
    unsigned char (*msk)[512] = sMask[l];
    const u16* Bg = a.wbf + OFFW[l];
    const float* bias = a.bias[l];
    prefetch_bytes(Bg, 32768, tid, 256);
    if (l == 3) {                       // 512 -> 256 (valid 253)
      v8f acc[2] = {};
      wmma_multi<512, 2>(As, 512, Bg, 256, wid * 16, 128, lane, acc);
#pragma unroll
      for (int t = 0; t < 2; ++t)
        store_leaky(acc[t], wid * 16 + t * 128 + r, hi, dst, msk, bias, Nv[l]);
    } else if (l == 0) {                // 288 -> 512
      v8f acc[4] = {};
      wmma_multi<288, 4>(As, 288, Bg, 512, wid * 16, 128, lane, acc);
#pragma unroll
      for (int t = 0; t < 4; ++t)
        store_leaky(acc[t], wid * 16 + t * 128 + r, hi, dst, msk, bias, Nv[l]);
    } else {                            // 512 -> 512
      v8f acc[4] = {};
      wmma_multi<512, 4>(As, 512, Bg, 512, wid * 16, 128, lane, acc);
#pragma unroll
      for (int t = 0; t < 4; ++t)
        store_leaky(acc[t], wid * 16 + t * 128 + r, hi, dst, msk, bias, Nv[l]);
    }
    __syncthreads();
    if (l == 3) {                       // latent_in: concat raw input (253..511)
      for (int c = c16; c < 259; c += 16) sAct[1][p16][253 + c] = sInp[p16][c];
      __syncthreads();
    }
  }

  // layer 8: s = act7 @ w8 + b8   (act7 in sAct[1])
  if (tid < 16) {
    float acc = a.b8[0];
    for (int k = 0; k < 512; ++k) acc += bf2f(sAct[1][tid][k]) * a.W8[k];
    a.out[S_OFF + (size_t)b * 2048 + nbase + tid] = acc;
  }
  // gp7 = w8 * mask7  -> sAct[0]
  for (int i = tid; i < 16 * 512; i += 256) {
    int p = i >> 9, k = i & 511;
    sAct[0][p][k] = f2bf(a.W8[k] * (sMask[7][p][k] ? 1.f : LEAKS));
  }
  __syncthreads();

  // ---------------- backward ----------------
  int cur = 0;
  for (int l = 7; l >= 1; --l) {
    const u16* As = &sAct[cur][0][0];
    const u16* Bg = a.wbf + OFFWT[l];
    u16 (*dst)[512] = sAct[cur ^ 1];
    prefetch_bytes(Bg, 32768, tid, 256);
    v8f acc[4] = {};
    if (l == 3) wmma_multi<256, 4>(As, 512, Bg, 512, wid * 16, 128, lane, acc);
    else        wmma_multi<512, 4>(As, 512, Bg, 512, wid * 16, 128, lane, acc);
#pragma unroll
    for (int t = 0; t < 4; ++t) {
      int col = wid * 16 + t * 128 + r;
#pragma unroll
      for (int j = 0; j < 8; ++j) {
        int m = hi * 8 + j;
        float g = acc[t][j];
        if (l == 4) {                   // split at latent_in concat
          if (col < 253) dst[m][col] = f2bf(g * (sMask[3][m][col] ? 1.f : LEAKS));
          else           sGin[m][col - 253] = g;   // grad wrt raw input
        } else {
          dst[m][col] = f2bf(g * (sMask[l - 1][m][col] ? 1.f : LEAKS));
        }
      }
    }
    if (l == 4 && tid < 48) { int p = tid / 3; dst[p][253 + tid % 3] = 0; }
    cur ^= 1;
    __syncthreads();
  }
  {                                     // l = 0: gin0 = gp0 @ W0^T (512x288)
    const u16* As = &sAct[cur][0][0];
    const u16* Bg = a.wbf + OFFWT[0];
    v8f acc[2] = {};
    wmma_multi<512, 2>(As, 512, Bg, 288, wid * 16, 128, lane, acc);
#pragma unroll
    for (int t = 0; t < 2; ++t) {
      int col = wid * 16 + t * 128 + r;           // 0..255
#pragma unroll
      for (int j = 0; j < 8; ++j) sGin[hi * 8 + j][col] += acc[t][j];
    }
    if (wid < 2) {                                // leftover cols 256..287
      v8f acc1[1] = {};
      wmma_multi<512, 1>(As, 512, Bg, 288, 256 + wid * 16, 128, lane, acc1);
      int col = 256 + wid * 16 + r;
#pragma unroll
      for (int j = 0; j < 8; ++j)
        if (col < 259) sGin[hi * 8 + j][col] += acc1[0][j];
    }
    __syncthreads();
  }

  // outputs: dsx (cols 256..258), dsh (sum over points of cols 0..255)
  if (tid < 48) {
    int p = tid / 3, d = tid % 3;
    a.out[DSX_OFF + (size_t)(b * 2048 + nbase + p) * 3 + d] = sGin[p][256 + d];
  }
  {
    float sum = 0.f;
#pragma unroll
    for (int p = 0; p < 16; ++p) sum += sGin[p][tid];
    atomicAdd(&a.out[DSH_OFF + b * 256 + tid], sum);
  }
}

// ---------------------------------------------------------------------------
// Kernel B: gated message passing.  1 WG = (b, 16-point n tile); loops m.
// z is stashed in the Gnew output slot; newG (max over m) kept in registers.
// ---------------------------------------------------------------------------
struct GateArgs {
  const float* X; const float* H; const float* P; const float* G;
  const u16* wbf;
  float* out;
  float* newG;
};

__global__ __launch_bounds__(256) void k_gate(GateArgs a) {
  __shared__ u16 sA[16][544];    // inp features (bf16, K padded to 544)
  __shared__ u16 sA2[16][544];   // inp2 features (cols 272.. replaced by r*G)

  const int tid = threadIdx.x, wid = tid >> 5, lane = tid & 31;
  const int r = lane & 15, hi = lane >> 4;
  const int wg = blockIdx.x, b = wg >> 7, nbase = (wg & 127) * 16;
  const int p16 = tid >> 4, c16 = tid & 15;

  const u16* Wr = a.wbf + OFF_WR;
  const u16* Wz = a.wbf + OFF_WZ;
  const u16* Wg = a.wbf + OFF_WG;
  prefetch_bytes(Wr, 544 * 256 * 2, tid, 256);
  prefetch_bytes(Wz, 544 * 256 * 2, tid, 256);
  prefetch_bytes(Wg, 544 * 256 * 2, tid, 256);

  // base features: [X(3) H(256) P(6)@259 s@265 dsx(3) dsx(3) G(256)@272, pad]
  {
    const int n = nbase + p16;
    for (int c = c16; c < 544; c += 16) {
      float v = 0.f;
      if (c < 3)        v = a.X[(size_t)(b * 2048 + n) * 3 + c];
      else if (c < 259) v = a.H[b * 256 + (c - 3)];
      else if (c < 265) v = 0.f;                                 // P per-m
      else if (c == 265) v = a.out[S_OFF + b * 2048 + n];        // s
      else if (c < 269) v = a.out[DSX_OFF + (size_t)(b * 2048 + n) * 3 + (c - 266)];
      else if (c < 272) v = a.out[DSX_OFF + (size_t)(b * 2048 + n) * 3 + (c - 269)];
      else if (c < 528) v = a.G[(size_t)(b * 2048 + n) * 256 + (c - 272)];
      u16 h = f2bf(v);
      sA[p16][c] = h;
      sA2[p16][c] = (c < 272) ? h : 0;
    }
  }
  __syncthreads();

  v8f vmax[2]; vmax[0] = {}; vmax[1] = {};      // relu outputs are >= 0

  for (int m = 0; m < 16; ++m) {
    if (tid < 96) {                             // refresh the 6 P features
      int p = tid / 6, c = tid % 6;
      u16 h = f2bf(a.P[(size_t)(b * 16 + m) * 6 + c]);
      sA[p][259 + c] = h; sA2[p][259 + c] = h;
    }
    __syncthreads();

    {  // r = relu(inp @ wr); write r*G into sA2 cols 272..527
      v8f acc[2] = {};
      wmma_multi<544, 2>(&sA[0][0], 544, Wr, 256, wid * 16, 128, lane, acc);
#pragma unroll
      for (int t = 0; t < 2; ++t) {
        int c = wid * 16 + t * 128 + r;
#pragma unroll
        for (int j = 0; j < 8; ++j) {
          int p = hi * 8 + j;
          float rv = fmaxf(acc[t][j], 0.f);
          float gv = a.G[(size_t)(b * 2048 + nbase + p) * 256 + c];
          sA2[p][272 + c] = f2bf(rv * gv);
        }
      }
    }
    {  // z = relu(inp @ wz) -> stash in the Gnew slot of d_out
      v8f acc[2] = {};
      wmma_multi<544, 2>(&sA[0][0], 544, Wz, 256, wid * 16, 128, lane, acc);
#pragma unroll
      for (int t = 0; t < 2; ++t) {
        int c = wid * 16 + t * 128 + r;
#pragma unroll
        for (int j = 0; j < 8; ++j) {
          int p = hi * 8 + j;
          a.out[GNEW_OFF + ((size_t)(b * 16 + m) * 2048 + nbase + p) * 256 + c] =
              fmaxf(acc[t][j], 0.f);
        }
      }
    }
    __syncthreads();
    {  // t = relu(inp2 @ wg); running max over m in registers
      v8f acc[2] = {};
      wmma_multi<544, 2>(&sA2[0][0], 544, Wg, 256, wid * 16, 128, lane, acc);
#pragma unroll
      for (int t = 0; t < 2; ++t)
#pragma unroll
        for (int j = 0; j < 8; ++j)
          vmax[t][j] = fmaxf(vmax[t][j], fmaxf(acc[t][j], 0.f));
    }
    __syncthreads();
  }

#pragma unroll
  for (int t = 0; t < 2; ++t) {
    int c = wid * 16 + t * 128 + r;
#pragma unroll
    for (int j = 0; j < 8; ++j) {
      int p = hi * 8 + j;
      a.newG[(size_t)(b * 2048 + nbase + p) * 256 + c] = vmax[t][j];
    }
  }
}

// ---------------------------------------------------------------------------
// Kernel C: Gnew = z*newG + (1-z)*G     (in-place over the z stash, float4)
// ---------------------------------------------------------------------------
__global__ __launch_bounds__(256) void k_combine(const float* __restrict__ G,
                                                 const float* __restrict__ newG,
                                                 float* out) {
  size_t i4 = ((size_t)blockIdx.x * 256 + threadIdx.x) * 4;   // 16 777 216 total
  int c = (int)(i4 & 255);
  size_t t = i4 >> 8;
  int n = (int)(t & 2047);
  int b = (int)(t >> 15);
  float4 z = *(const float4*)(out + GNEW_OFF + i4);
  size_t gi = ((size_t)(b * 2048 + n)) * 256 + c;
  float4 ng = *(const float4*)(newG + gi);
  float4 g  = *(const float4*)(G + gi);
  float4 o;
  o.x = z.x * ng.x + (1.f - z.x) * g.x;
  o.y = z.y * ng.y + (1.f - z.y) * g.y;
  o.z = z.z * ng.z + (1.f - z.z) * g.z;
  o.w = z.w * ng.w + (1.f - z.w) * g.w;
  *(float4*)(out + GNEW_OFF + i4) = o;
}

// ---------------------------------------------------------------------------
// Kernel D: heads.  lp per row, dx = global max per b (int atomicMax trick).
// ---------------------------------------------------------------------------
__global__ __launch_bounds__(256) void k_heads(const float* __restrict__ X,
                                               const float* __restrict__ H,
                                               const float* __restrict__ P,
                                               const float* __restrict__ wdx,
                                               const float* __restrict__ wlp,
                                               float* out) {
  int row = blockIdx.x * 256 + threadIdx.x;          // 65536 rows (b,m,n)
  int n = row & 2047, bm = row >> 11, b = bm >> 4, m = bm & 15;
  float a1 = 0.f, a2 = 0.f;
  int k = 0;
  const float* xq = X + (size_t)(b * 2048 + n) * 3;
  for (int d = 0; d < 3; ++d, ++k) { float f = xq[d]; a1 += f * wdx[k]; a2 += f * wlp[k]; }
  for (int d = 0; d < 256; ++d, ++k) { float f = H[b * 256 + d]; a1 += f * wdx[k]; a2 += f * wlp[k]; }
  for (int d = 0; d < 6; ++d, ++k) { float f = P[(size_t)(b * 16 + m) * 6 + d]; a1 += f * wdx[k]; a2 += f * wlp[k]; }
  { float f = out[S_OFF + b * 2048 + n]; a1 += f * wdx[k]; a2 += f * wlp[k]; ++k; }
  const float* dq = out + DSX_OFF + (size_t)(b * 2048 + n) * 3;
  for (int d = 0; d < 3; ++d, ++k) { float f = dq[d]; a1 += f * wdx[k]; a2 += f * wlp[k]; }
  for (int d = 0; d < 3; ++d, ++k) { float f = dq[d]; a1 += f * wdx[k]; a2 += f * wlp[k]; }
  const float* gq = out + GNEW_OFF + (size_t)row * 256;   // k == 272 here
  for (int d = 0; d < 256; d += 4) {
    float4 f  = *(const float4*)(gq + d);
    float4 wa = *(const float4*)(wdx + 272 + d);
    float4 wb = *(const float4*)(wlp + 272 + d);
    a1 += f.x * wa.x + f.y * wa.y + f.z * wa.z + f.w * wa.w;
    a2 += f.x * wb.x + f.y * wb.y + f.z * wb.z + f.w * wb.w;
  }
  out[LP_OFF + row] = fmaxf(a2, 0.f);
  float dxv = fmaxf(a1, 0.f);                        // >= 0 so int-max is valid
  atomicMax((int*)&out[DX_OFF + b], __float_as_int(dxv));
}

// ---------------------------------------------------------------------------
// Host launcher
// ---------------------------------------------------------------------------
extern "C" void kernel_launch(void* const* d_in, const int* in_sizes, int n_in,
                              void* d_out, int out_size, void* d_ws, size_t ws_size,
                              hipStream_t stream) {
  const float* X = (const float*)d_in[0];
  const float* H = (const float*)d_in[1];
  const float* P = (const float*)d_in[2];
  const float* G = (const float*)d_in[3];
  // params: sdf_w{i}=d_in[4+2i], sdf_b{i}=d_in[5+2i]; wr=22 wz=23 wg=24 wdx=25 wlp=26
  u16* wbf = (u16*)d_ws;
  float* newG = (float*)((char*)d_ws + WS_BF16_END_BYTES);
  float* out = (float*)d_out;

  hipMemsetAsync(out + DSH_OFF, 0, 512 * sizeof(float), stream);  // dsh accum
  hipMemsetAsync(out + DX_OFF, 0, 2 * sizeof(float), stream);     // dx max

  static const int IN_[8]   = {259,512,512,512,512,512,512,512};
  static const int OUT_[8]  = {512,512,512,253,512,512,512,512};
  static const int INp_[8]  = {288,512,512,512,512,512,512,512};
  static const int OUTp_[8] = {512,512,512,256,512,512,512,512};
  static const int hOFFW[8]  = {0, 147456, 409600, 671744, 802816, 1064960, 1327104, 1589248};
  static const int hOFFWT[8] = {1851392, 1998848, 2260992, 2523136, 2654208, 2916352, 3178496, 3440640};
  for (int l = 0; l < 8; ++l)
    k_conv_sdf<<<512, 256, 0, stream>>>((const float*)d_in[4 + 2 * l],
                                        wbf + hOFFW[l], wbf + hOFFWT[l],
                                        IN_[l], OUT_[l], INp_[l], OUTp_[l]);
  k_conv_gate<<<272, 256, 0, stream>>>((const float*)d_in[22], wbf + OFF_WR, 528, 256, 544);
  k_conv_gate<<<272, 256, 0, stream>>>((const float*)d_in[23], wbf + OFF_WZ, 528, 256, 544);
  k_conv_gate<<<272, 256, 0, stream>>>((const float*)d_in[24], wbf + OFF_WG, 528, 256, 544);

  SdfArgs sa;
  sa.X = X; sa.H = H;
  sa.W8 = (const float*)d_in[20]; sa.b8 = (const float*)d_in[21];
  for (int l = 0; l < 8; ++l) sa.bias[l] = (const float*)d_in[5 + 2 * l];
  sa.wbf = wbf; sa.out = out;
  k_sdf<<<256, 256, 0, stream>>>(sa);

  GateArgs ga{X, H, P, G, wbf, out, newG};
  k_gate<<<256, 256, 0, stream>>>(ga);

  k_combine<<<16384, 256, 0, stream>>>(G, newG, out);

  k_heads<<<256, 256, 0, stream>>>(X, H, P, (const float*)d_in[25],
                                   (const float*)d_in[26], out);
}
